// GarchNet_7559142441150
// MI455X (gfx1250) — matrix operand order repaired and checked by reading, not compiled, not run on hardware
//
#include <hip/hip_runtime.h>
#include <math.h>

typedef __attribute__((ext_vector_type(16))) _Float16 v16h;
typedef __attribute__((ext_vector_type(8)))  float    v8f;

#define TT      256     // output times per block
#define CH      16      // channels per block (one per wave32)
#define NTH     512     // 16 waves
#define Y_HALO  112     // max lookback: 16*Qf(<=96) + rem(<=15) + 1
#define Y_LEN   368     // Y_HALO + TT
#define Y_PAD   369     // odd stride -> conflict-free banks
#define S_HALO  80      // 16*P0 (P0 <= 5)
#define S_LEN   336     // S_HALO + TT
#define S_PAD   337
#define O_PAD   17
#define SCALE     4096.0f
#define INV_SCALE (1.0f/4096.0f)

__device__ __forceinline__ float sigmoidf_(float v) { return 1.0f / (1.0f + expf(-v)); }

// ---- Kernel 1: device-side num_weights (graph-capture safe, no host readback) ----
__global__ void garch_nw_kernel(const float* __restrict__ beta_raw, int N,
                                int* __restrict__ nw_out) {
  __shared__ float red[256];
  float m = -3.4e38f;
  for (int i = threadIdx.x; i < N; i += 256) m = fmaxf(m, beta_raw[i]);
  red[threadIdx.x] = m;
  __syncthreads();
  for (int s = 128; s > 0; s >>= 1) {
    if (threadIdx.x < s) red[threadIdx.x] = fmaxf(red[threadIdx.x], red[threadIdx.x + s]);
    __syncthreads();
  }
  if (threadIdx.x == 0) {
    float bmax = sigmoidf_(red[0]) * 0.9999f;            // sigmoid monotonic -> max(raw)
    double nwd = ceil(log(0.01) / log((double)bmax)) + 1.0;
    int nw = (int)nwd;
    if (nw > 100) nw = 100;
    if (nw < 1)  nw = 1;
    nw_out[0] = nw;
  }
}

// ---- Kernel 2: async-staged, WMMA-combined GARCH FIR ----
__global__ __launch_bounds__(NTH) void garch_fir_wmma_kernel(
    const float* __restrict__ x,
    const float* __restrict__ alpha_raw,
    const float* __restrict__ beta_raw,
    const float* __restrict__ omega_raw,
    const int*   __restrict__ nw_ptr,
    float*       __restrict__ out,
    int T, int N)
{
  __shared__ float y_lds[CH][Y_PAD];     // raw x, [channel][time] (async-gathered transpose)
  __shared__ float s_lds[CH][S_PAD];     // 16-tap prefilter S (unscaled)
  __shared__ float p16_lds[CH][8];       // beta^(16e) table
  __shared__ float o_lds[TT * O_PAD];    // output transpose tile

  const int q  = threadIdx.x;
  const int c0 = blockIdx.x * CH;
  const int t0 = blockIdx.y * TT;

  // Phase 1: async global -> LDS transposing gather of the x tile.
  // Each lane supplies its own LDS dest (= transposed slot) and global address;
  // the async engine moves cache->LDS without touching VGPRs.
  {
    const int cc  = q & (CH - 1);
    const int r0  = q >> 4;                               // 0..31
    const int nch = (c0 + cc < N) ? (c0 + cc) : (N - 1);  // clamp; stores guarded later
    for (int Lr = r0; Lr < Y_LEN; Lr += NTH / CH) {
      int t = t0 - Y_HALO + Lr;
      if (t >= 0 && t < T) {
        unsigned lds_addr = (unsigned)(size_t)&y_lds[cc][Lr];   // LDS byte offset (addr[31:0])
        const float* gptr = x + (size_t)t * N + nch;
        asm volatile("global_load_async_to_lds_b32 %0, %1, off"
                     :: "v"(lds_addr), "v"(gptr) : "memory");
      } else {
        y_lds[cc][Lr] = 0.0f;                              // zero-pad t<0 halo
      }
    }
    asm volatile("s_wait_asynccnt 0x0" ::: "memory");      // wave's async loads done
  }
  __syncthreads();

  // Phase 2: per-wave (per-channel) parameters
  const int wave = q >> 5;
  const int lane = q & 31;
  const int cc   = wave;
  const int n    = (c0 + cc < N) ? (c0 + cc) : (N - 1);

  const float alpha = sigmoidf_(alpha_raw[n]);
  const float beta  = sigmoidf_(beta_raw[n]) * 0.9999f;
  const float omg   = sigmoidf_(omega_raw[n]);
  const float w0    = (omg * omg) / (1.0f - beta);

  const int nw  = nw_ptr[0];
  const int Qf  = nw >> 4;       // full 16-lag blocks (<= 6 since nw <= 100)
  const int rem = nw & 15;       // remainder taps
  const int P0  = Qf - 1;

  const float b2 = beta * beta, b4 = b2 * b2, b8 = b4 * b4;
  const float b16 = b8 * b8;
  float bQ = 1.0f;
  for (int s = 0; s < Qf; ++s) bQ *= b16;   // beta^(16*Qf)

  if (lane < 8) {
    float p = 1.0f;
    for (int s = 0; s < lane; ++s) p *= b16;
    p16_lds[cc][lane] = p;                  // beta^(16*lane)
  }

  // Phase 3: S[t] = sum_{r=1..16} beta^(r-1) * x^2[t-r]  (per-wave, own LDS row)
  for (int Ls = lane; Ls < S_LEN; Ls += 32) {
    float acc = 0.0f, w = 1.0f;
#pragma unroll
    for (int r = 1; r <= 16; ++r) {
      float xv = y_lds[cc][Ls + 32 - r];    // y idx = (t0-80+Ls) - r + Y_HALO
      acc += w * (xv * xv);
      w *= beta;
    }
    s_lds[cc][Ls] = acc;
  }

  // Phase 4: build WMMA operands per ISA lane layouts
  const int mj   = lane & 15;    // A: row i ; B: col j ; D: col j
  const int half = lane >> 4;

  v16h Aop, Bop;
#pragma unroll
  for (int v = 0; v < 8; ++v) {
    const int KsA = (v < 4) ? (2 * v + 8 * half) : (16 + 2 * (v - 4) + 8 * half);
    const int KsB = (v < 4) ? (2 * v + 8 * half) : (16 + 2 * (v - 4) + 8 * half);
#pragma unroll
    for (int u = 0; u < 2; ++u) {
      // A[i,k] = SCALE * S[t0 + i + 16*(k - P0)]
      int ka = KsA + u;
      float sv = 0.0f;
      if (ka <= 15 + P0) {
        int Ls = mj + 16 * (ka - P0) + S_HALO;
        if (Ls >= 0 && Ls < S_LEN) sv = s_lds[cc][Ls] * SCALE;
      }
      Aop[2 * v + u] = (_Float16)sv;
      // B[k,j] = beta^(16*(j-k+P0)) if 0 <= j-k+P0 < Qf else 0  (banded)
      int kb = KsB + u;
      int e  = mj - kb + P0;
      float bv = 0.0f;
      if (e >= 0 && e < Qf) bv = p16_lds[cc][e];
      Bop[2 * v + u] = (_Float16)bv;
    }
  }

  // Phase 5: one WMMA per 16x16 time tile (256 outputs), f32 accumulate
  v8f cacc = {};
  cacc = __builtin_amdgcn_wmma_f32_16x16x32_f16(
      /*neg_a=*/false, Aop, /*neg_b=*/false, Bop,
      /*c_mod=*/(short)0, cacc, /*reuse_a=*/false, /*reuse_b=*/false);

  // Phase 6: remainder taps + epilogue -> padded LDS transpose
#pragma unroll
  for (int r = 0; r < 8; ++r) {
    const int i_out = r + 8 * half;          // D row
    const int t_l   = i_out + 16 * mj;       // local time 0..255
    float srem = 0.0f;
    if (rem > 0) {
      int Lyb = t_l - 16 * Qf + Y_HALO;      // y idx for lag 16*Qf
      float w = 1.0f;
      for (int rr = 1; rr <= rem; ++rr) {
        float xv = y_lds[cc][Lyb - rr];
        srem += w * (xv * xv);
        w *= beta;
      }
    }
    float a   = cacc[r] * INV_SCALE + bQ * srem;
    float var = w0 + alpha * a;
    o_lds[t_l * O_PAD + cc] = sqrtf(var);
  }
  __syncthreads();

  // Phase 7: async LDS -> global scatter of the transposed output tile
  // (16 contiguous floats per output row; no VGPR round trip)
  {
    const int ccs = q & (CH - 1);
    const int r0  = q >> 4;
    const int nch = c0 + ccs;
    if (nch < N) {
      for (int t_l = r0; t_l < TT; t_l += NTH / CH) {
        int t = t0 + t_l;
        if (t < T) {
          unsigned lds_addr = (unsigned)(size_t)&o_lds[t_l * O_PAD + ccs];
          float* gptr = out + (size_t)t * N + nch;
          asm volatile("global_store_async_from_lds_b32 %0, %1, off"
                       :: "v"(gptr), "v"(lds_addr) : "memory");
        }
      }
    }
    asm volatile("s_wait_asynccnt 0x0" ::: "memory");      // drain before wave exit
  }
}

extern "C" void kernel_launch(void* const* d_in, const int* in_sizes, int n_in,
                              void* d_out, int out_size, void* d_ws, size_t ws_size,
                              hipStream_t stream) {
  const float* x         = (const float*)d_in[0];
  const float* alpha_raw = (const float*)d_in[1];
  const float* beta_raw  = (const float*)d_in[2];
  const float* omega_raw = (const float*)d_in[3];
  float* out = (float*)d_out;

  const int N = in_sizes[3];           // omega_raw has shape [N]
  const int T = in_sizes[0] / N;       // x has shape [T, N]

  int* nw_ptr = (int*)d_ws;

  garch_nw_kernel<<<1, 256, 0, stream>>>(beta_raw, N, nw_ptr);

  dim3 grid((N + CH - 1) / CH, (T + TT - 1) / TT);
  garch_fir_wmma_kernel<<<grid, NTH, 0, stream>>>(
      x, alpha_raw, beta_raw, omega_raw, nw_ptr, out, T, N);
}
